// EEGCapsNet_13477607375371
// MI455X (gfx1250) — matrix operand order repaired and checked by reading, not compiled, not run on hardware
//
#include <hip/hip_runtime.h>
#include <hip/hip_bf16.h>

// ---------------------------------------------------------------------------
// EEG CapsNet forward for gfx1250 (MI455X).
//  - conv branches: implicit-im2col WMMA (bf16), offset-table gather,
//    2 passes (BatchNorm batch statistics in between)
//  - dynamic routing: templated wave32 kernels (power-of-2 shapes), LDS fp32
//  - reconstruction MLPs: 128x64 block-tile WMMA GEMM, 4 wmma/wave/K-step,
//    A tile filled with global_load_async_to_lds_b128 (+ s_wait_asynccnt)
// ---------------------------------------------------------------------------

typedef __attribute__((ext_vector_type(16))) __bf16   v16bf;
typedef __attribute__((ext_vector_type(8)))  float    v8f;
typedef __attribute__((ext_vector_type(8)))  unsigned u32x8;

__device__ __forceinline__ unsigned short bfbits(float f) {
    unsigned u = __builtin_bit_cast(unsigned, f);
    return (unsigned short)((u + 0x7fffu + ((u >> 16) & 1u)) >> 16);   // RNE
}
__device__ __forceinline__ __bf16 f2bf(float f) {
    unsigned short h = bfbits(f);
    return __builtin_bit_cast(__bf16, h);
}
__device__ __forceinline__ unsigned packbf(float lo, float hi) {
    return (unsigned)bfbits(lo) | ((unsigned)bfbits(hi) << 16);
}
__device__ __forceinline__ unsigned lds_off(const void* p) {
    // flat shared aperture: addr[31:0] is the LDS offset (ISA 10.2 aperture rules)
    return (unsigned)(unsigned long long)p;
}
__device__ __forceinline__ float wave_sum(float v) {
#pragma unroll
    for (int m = 16; m > 0; m >>= 1) v += __shfl_xor(v, m, 32);
    return v;
}

// -------------------------------- utility ----------------------------------
__global__ void zero_kernel(float* p, int n) {
    for (int i = blockIdx.x * blockDim.x + threadIdx.x; i < n; i += gridDim.x * blockDim.x)
        p[i] = 0.f;
}
__global__ void f32_to_bf16_kernel(const float* __restrict__ s, __bf16* __restrict__ d, int n) {
    for (int i = blockIdx.x * blockDim.x + threadIdx.x; i < n; i += gridDim.x * blockDim.x)
        d[i] = f2bf(s[i]);
}

// ---------------- conv branch: implicit-im2col WMMA core -------------------
// Per block: one (b,g).  sx: H*192 fp32 input rows, zero padded to 2048
// (region [1792,2048) is the zero sentinel target).  wt: 16 x Kpad bf16
// weight tile, transposed [n][k].  soff2[k/2] = {off(k), off(k+1)} where
// off(k) = r*192+kw for k<Kreal else 1792 (reads zero).  4 waves; wave wv
// owns position tiles {wv, wv+4, wv+8}.
__device__ __forceinline__ void conv_wmma_compute(const float* sx, const __bf16* wt,
                                                  const int2* soff2, float* sh,
                                                  const float* b1, int g,
                                                  int Wout, int Kpad, int ntiles) {
    const int tid = threadIdx.x, lane = tid & 31, wv = tid >> 5;
    const int hi = lane >> 4, l15 = lane & 15;
    const float b1v = b1[g * 8 + (l15 & 7)];
    const int nt_w = (ntiles - wv + 3) >> 2;        // tiles for this wave (<=3)

    v8f acc[3];
#pragma unroll
    for (int j = 0; j < 3; ++j) acc[j] = (v8f){};

    for (int kb = 0; kb < Kpad; kb += 32) {
        u32x8 bu;                                    // B fragment: per-kb only
#pragma unroll
        for (int v = 0; v < 8; ++v)
            bu[v] = *(const unsigned*)&wt[l15 * Kpad + kb + ((v >= 4) ? 16 : 0) + (v & 3) * 2 + hi * 8];
        v16bf bf = __builtin_bit_cast(v16bf, bu);
#pragma unroll
        for (int j = 0; j < 3; ++j) {
            if (j < nt_w) {                          // uniform per wave
                const int pos = (wv + 4 * j) * 16 + l15;
                u32x8 au;
#pragma unroll
                for (int v = 0; v < 8; ++v) {
                    int k0 = kb + ((v >= 4) ? 16 : 0) + (v & 3) * 2 + hi * 8;
                    int2 o2 = soff2[k0 >> 1];
                    au[v] = packbf(sx[o2.x + pos], sx[o2.y + pos]);
                }
                acc[j] = __builtin_amdgcn_wmma_f32_16x16x32_bf16(
                    false, __builtin_bit_cast(v16bf, au), false, bf,
                    (short)0, acc[j], false, false);
            }
        }
    }
#pragma unroll
    for (int j = 0; j < 3; ++j)
        if (j < nt_w && l15 < 8) {
#pragma unroll
            for (int v = 0; v < 8; ++v) {
                int p = (wv + 4 * j) * 16 + v + hi * 8;
                if (p < Wout) sh[l15 * 192 + p] = acc[j][v] + b1v;
            }
        }
}

__device__ __forceinline__ void conv_load_tiles(const float* __restrict__ x,
                                                const float* __restrict__ w1,
                                                float* sx, __bf16* wt, int2* soff2,
                                                int b, int g, int H, int KW,
                                                int Kreal, int Kpad) {
    const int tid = threadIdx.x;
    const float* xin = x + ((size_t)b * 16 + (size_t)g * H) * 192;
    for (int i = tid; i < H * 192; i += 128) sx[i] = xin[i];
    for (int i = H * 192 + tid; i < 2048; i += 128) sx[i] = 0.f;
    const float* wg = w1 + (size_t)(g * 8) * Kreal;
    for (int n = 0; n < 16; ++n)
        for (int k = tid; k < Kpad; k += 128)
            wt[n * Kpad + k] = (n < 8 && k < Kreal) ? f2bf(wg[n * Kreal + k]) : f2bf(0.f);
    for (int k2 = tid; k2 < (Kpad >> 1); k2 += 128) {
        int k = 2 * k2;
        int r0 = k / KW,       o0 = (k < Kreal)     ? r0 * 192 + (k - r0 * KW)           : 1792;
        int r1 = (k + 1) / KW, o1 = (k + 1 < Kreal) ? r1 * 192 + ((k + 1) - r1 * KW)     : 1792;
        soff2[k2] = make_int2(o0, o1);
    }
}

// pass 1: per-channel sum/sumsq over (b,pos) for BatchNorm batch stats
__global__ void conv_stats_kernel(const float* __restrict__ x, const float* __restrict__ w1,
                                  const float* __restrict__ b1,
                                  float* __restrict__ sum_, float* __restrict__ sumsq_,
                                  int G, int H, int KW, int Wout, int Kreal, int Kpad,
                                  int ntiles) {
    __shared__ float  sx[2048];
    __shared__ __bf16 wt[8192];
    __shared__ float  sh[1536];
    __shared__ int2   soff2[256];
    __shared__ float  cst[16];
    const int b = blockIdx.x, g = blockIdx.y, tid = threadIdx.x, lane = tid & 31;
    conv_load_tiles(x, w1, sx, wt, soff2, b, g, H, KW, Kreal, Kpad);
    __syncthreads();
    conv_wmma_compute(sx, wt, soff2, sh, b1, g, Wout, Kpad, ntiles);
    if (tid < 16) cst[tid] = 0.f;
    __syncthreads();

    float sacc[8], qacc[8];
#pragma unroll
    for (int c = 0; c < 8; ++c) { sacc[c] = 0.f; qacc[c] = 0.f; }
    for (int pos = tid; pos < Wout; pos += 128)
#pragma unroll
        for (int c = 0; c < 8; ++c) {
            float h = sh[c * 192 + pos];
            sacc[c] += h; qacc[c] += h * h;
        }
#pragma unroll
    for (int c = 0; c < 8; ++c) {
        float s = wave_sum(sacc[c]);
        float q = wave_sum(qacc[c]);
        if (lane == 0) { atomicAdd(&cst[c], s); atomicAdd(&cst[8 + c], q); }
    }
    __syncthreads();
    if (tid < 8) {
        atomicAdd(&sum_[g * 8 + tid],   cst[tid]);
        atomicAdd(&sumsq_[g * 8 + tid], cst[8 + tid]);
    }
}

__global__ void bn_finalize_kernel(const float* sum_, const float* sumsq_,
                                   const float* gamma, const float* beta,
                                   float* scale, float* shift, int C, float invCount) {
    int c = blockIdx.x * blockDim.x + threadIdx.x;
    if (c < C) {
        float mu  = sum_[c] * invCount;
        float var = sumsq_[c] * invCount - mu * mu;
        float sc  = gamma[c] * rsqrtf(var + 1e-5f);
        scale[c] = sc;
        shift[c] = beta[c] - mu * sc;
    }
}

// pass 2: recompute conv1 (WMMA) + bn + act + conv2 dot over Wout
// act: 0=relu 1=elu.  Writes 16 vals per (b,g) into uout(B,G,32)+col_off.
__global__ void conv2_kernel(const float* __restrict__ x, const float* __restrict__ w1,
                             const float* __restrict__ b1,
                             const float* __restrict__ scale, const float* __restrict__ shift,
                             const float* __restrict__ w2, const float* __restrict__ b2,
                             float* __restrict__ uout,
                             int G, int H, int KW, int Wout, int Kreal, int Kpad,
                             int ntiles, int act, int col_off) {
    __shared__ float  sx[2048];
    __shared__ __bf16 wt[8192];
    __shared__ float  sh[1536];
    __shared__ int2   soff2[256];
    __shared__ float  cac[16];
    const int b = blockIdx.x, g = blockIdx.y, tid = threadIdx.x, lane = tid & 31;
    conv_load_tiles(x, w1, sx, wt, soff2, b, g, H, KW, Kreal, Kpad);
    __syncthreads();
    conv_wmma_compute(sx, wt, soff2, sh, b1, g, Wout, Kpad, ntiles);
    __syncthreads();

    for (int i = tid; i < 8 * Wout; i += 128) {        // batchnorm + activation
        int c = i / Wout, p = i - c * Wout;
        float v = sh[c * 192 + p] * scale[g * 8 + c] + shift[g * 8 + c];
        sh[c * 192 + p] = act ? (v > 0.f ? v : (__expf(v) - 1.f)) : fmaxf(v, 0.f);
    }
    if (tid < 16) cac[tid] = 0.f;
    __syncthreads();

    float acc[16];
#pragma unroll
    for (int j = 0; j < 16; ++j) acc[j] = 0.f;
    for (int pos = tid; pos < Wout; pos += 128)
#pragma unroll
        for (int c = 0; c < 8; ++c) {
            float hv = sh[c * 192 + pos];
            acc[2 * c]     = fmaf(hv, w2[(size_t)(g * 16 + 2 * c) * Wout + pos],     acc[2 * c]);
            acc[2 * c + 1] = fmaf(hv, w2[(size_t)(g * 16 + 2 * c + 1) * Wout + pos], acc[2 * c + 1]);
        }
#pragma unroll
    for (int j = 0; j < 16; ++j) {
        float s = wave_sum(acc[j]);
        if (lane == 0) atomicAdd(&cac[j], s);
    }
    __syncthreads();
    if (tid < 16)
        uout[((size_t)b * G + g) * 32 + col_off + tid] = cac[tid] + b2[g * 16 + tid];
}

// ----------------------------- dynamic routing -----------------------------
// Templated on power-of-2 shapes -> all div/mod become shifts, loops unroll.
template <int I, int O, int D, int E>
__global__ void routing_kernel(const float* __restrict__ u, const float* __restrict__ W,
                               float* __restrict__ vout, int G) {
    __shared__ float uh[I * O * E];
    __shared__ float ub[I * D];
    __shared__ float bb[I * O];
    __shared__ float cc[I * O];
    __shared__ float sv[O * E];
    __shared__ float vv[O * E];
    const int bg = blockIdx.x, g = bg % G, tid = threadIdx.x;

    for (int i = tid; i < I * D; i += 32) ub[i] = u[(size_t)bg * I * D + i];
    for (int i = tid; i < I * O; i += 32) bb[i] = 0.f;
    __syncthreads();

    const float* Wg = W + (size_t)g * I * O * D * E;
    for (int idx = tid; idx < I * O * E; idx += 32) {
        int i = idx / (O * E), o = (idx / E) % O, e = idx % E;
        float acc = 0.f;
#pragma unroll
        for (int d = 0; d < D; ++d)
            acc = fmaf(ub[i * D + d], Wg[((size_t)(i * O + o) * D + d) * E + e], acc);
        uh[idx] = acc;
    }
    __syncthreads();

    for (int iter = 0; iter < 3; ++iter) {
        for (int i = tid; i < I; i += 32) {            // softmax over O per i
            float mx = -3.4e38f;
#pragma unroll
            for (int o = 0; o < O; ++o) mx = fmaxf(mx, bb[i * O + o]);
            float sm = 0.f;
#pragma unroll
            for (int o = 0; o < O; ++o) { float ex = __expf(bb[i * O + o] - mx); cc[i * O + o] = ex; sm += ex; }
            float inv = 1.f / sm;
#pragma unroll
            for (int o = 0; o < O; ++o) cc[i * O + o] *= inv;
        }
        __syncthreads();
        for (int idx = tid; idx < O * E; idx += 32) {  // s = sum_i c * u_hat
            int o = idx / E, e = idx % E;
            float acc = 0.f;
#pragma unroll
            for (int i = 0; i < I; ++i) acc = fmaf(cc[i * O + o], uh[(i * O + o) * E + e], acc);
            sv[idx] = acc;
        }
        __syncthreads();
        for (int o = tid; o < O; o += 32) {            // squash
            float n2 = 0.f;
#pragma unroll
            for (int e = 0; e < E; ++e) n2 += sv[o * E + e] * sv[o * E + e];
            float f = n2 / ((1.f + n2) * sqrtf(n2 + 1e-8f));
#pragma unroll
            for (int e = 0; e < E; ++e) vv[o * E + e] = sv[o * E + e] * f;
        }
        __syncthreads();
        if (iter < 2) {                                // b += u_hat . v
            for (int idx = tid; idx < I * O; idx += 32) {
                int i = idx / O, o = idx % O;
                float acc = 0.f;
#pragma unroll
                for (int e = 0; e < E; ++e) acc = fmaf(uh[(i * O + o) * E + e], vv[o * E + e], acc);
                bb[idx] += acc;
            }
            __syncthreads();
        }
    }
    for (int idx = tid; idx < O * E; idx += 32)
        vout[(size_t)bg * O * E + idx] = vv[idx];
}

// ------------------------------- small helpers -----------------------------
__global__ void concat_caps_kernel(const float* a, const float* b_, float* out,
                                   int BG, int C, int E) {
    int idx = blockIdx.x * blockDim.x + threadIdx.x;
    int total = BG * 2 * C * E;
    if (idx >= total) return;
    int e = idx % E, c = (idx / E) % (2 * C), bg = idx / (E * 2 * C);
    out[idx] = (c < C) ? a[((size_t)bg * C + c) * E + e] : b_[((size_t)bg * C + (c - C)) * E + e];
}

__global__ void mask_bf16_kernel(const float* v, const int* y, __bf16* m, int B, int E) {
    int idx = blockIdx.x * blockDim.x + threadIdx.x;
    if (idx >= B * 2 * E) return;
    int cls = (idx / E) % 2, b = idx / (2 * E);
    m[idx] = f2bf((y[b] == cls) ? v[idx] : 0.f);
}

__global__ void capslen_kernel(const float* v, float* out, int B, int NC, int E) {
    int idx = blockIdx.x * blockDim.x + threadIdx.x;
    if (idx >= B * NC) return;
    const float* p = v + (size_t)idx * E;
    float s = 0.f;
    for (int e = 0; e < E; ++e) s += p[e] * p[e];
    out[idx] = sqrtf(s + 1e-8f);
}

// ------------------- WMMA GEMM: C = act(A*W + bias), bf16 ------------------
#define BM 128
#define BN 64
#define BK 32
#define LDA 40   // bf16 elems/row: 80B rows (16B-aligned), conflict-free stride
#define LDB 40

__global__ void gemm_bf16_kernel(const __bf16* __restrict__ A, const __bf16* __restrict__ W,
                                 const float* __restrict__ bias, void* __restrict__ Cout,
                                 int M, int N, int K, int act, int out_bf16) {
    __shared__ __bf16 As[BM * LDA];    // [m][k]
    __shared__ __bf16 Bt[BN * LDB];    // [n][k] (transposed)
    const int tid = threadIdx.x, lane = tid & 31, wv = tid >> 5;
    const int hi = lane >> 4, l15 = lane & 15;
    const int bm = blockIdx.x * BM, bn = blockIdx.y * BN;
    const int mw = (wv >> 1) * 32, nw = (wv & 1) * 32;

    const unsigned asbase = lds_off(&As[0]);
    const unsigned long long abase = (unsigned long long)A;

    v8f acc[2][2];
#pragma unroll
    for (int i = 0; i < 2; ++i)
#pragma unroll
        for (int j = 0; j < 2; ++j) acc[i][j] = (v8f){};

    for (int kb = 0; kb < K; kb += BK) {
        // --- A tile: 128 rows x 64B via async b128 copies (2 chunks/thread) ---
#pragma unroll
        for (int c = 0; c < 2; ++c) {
            int chunk = tid + c * 256;                 // 512 chunks
            int row = chunk >> 2, c4 = chunk & 3;
            int gm = bm + row;
            if (gm < M) {
                unsigned voff = (unsigned)(((size_t)gm * K + kb) * 2 + c4 * 16);
                unsigned ldst = asbase + row * (LDA * 2) + c4 * 16;
                asm volatile("global_load_async_to_lds_b128 %0, %1, %2"
                             :: "v"(ldst), "v"(voff), "s"(abase) : "memory");
            }
        }
        // --- B tile transposed: Bt[n][k] = W[kb+k][bn+n] ---
        for (int i = tid; i < BK * BN; i += 256) {
            int n = i & 63, k = i >> 6;
            Bt[n * LDB + k] = W[(size_t)(kb + k) * N + bn + n];
        }
        if (kb + BK < K)
            __builtin_prefetch(W + (size_t)(kb + BK + (tid & 31)) * N + bn, 0, 1);
        asm volatile("s_wait_asynccnt 0x0" ::: "memory");
        __syncthreads();

        v16bf af[2], bf[2];
#pragma unroll
        for (int t = 0; t < 2; ++t) {
            u32x8 au, bu;
            const int mrow = mw + t * 16 + l15;
            const int ncol = nw + t * 16 + l15;
#pragma unroll
            for (int v = 0; v < 8; ++v) {
                int k0 = ((v >= 4) ? 16 : 0) + (v & 3) * 2 + hi * 8;
                au[v] = *(const unsigned*)&As[mrow * LDA + k0];
                bu[v] = *(const unsigned*)&Bt[ncol * LDB + k0];
            }
            af[t] = __builtin_bit_cast(v16bf, au);
            bf[t] = __builtin_bit_cast(v16bf, bu);
        }
#pragma unroll
        for (int i = 0; i < 2; ++i)
#pragma unroll
            for (int j = 0; j < 2; ++j)
                acc[i][j] = __builtin_amdgcn_wmma_f32_16x16x32_bf16(
                    false, af[i], false, bf[j], (short)0, acc[i][j], false, false);
        __syncthreads();
    }

#pragma unroll
    for (int i = 0; i < 2; ++i)
#pragma unroll
        for (int j = 0; j < 2; ++j) {
            int gn = bn + nw + j * 16 + l15;
            float bv = bias[gn];
#pragma unroll
            for (int v = 0; v < 8; ++v) {
                int gm = bm + mw + i * 16 + v + hi * 8;
                float val = acc[i][j][v] + bv;
                if (act == 1)      val = fmaxf(val, 0.f);
                else if (act == 2) val = 1.f / (1.f + __expf(-val));
                if (gm < M) {
                    if (out_bf16) ((__bf16*)Cout)[(size_t)gm * N + gn] = f2bf(val);
                    else          ((float*)Cout)[(size_t)gm * N + gn]  = val;
                }
            }
        }
}

// =============================== host side =================================
extern "C" void kernel_launch(void* const* d_in, const int* in_sizes, int n_in,
                              void* d_out, int out_size, void* d_ws, size_t ws_size,
                              hipStream_t stream) {
    const float* x = (const float*)d_in[0];
    const int*   y = (const int*)d_in[1];
    const int B = in_sizes[1];
    float* dout = (float*)d_out;

    auto F = [&](int i) { return (const float*)d_in[i]; };
    // params flattened (nested dicts sorted by key):
    // 2 channelDeepSuper, 3 channelRouting, 4 channelShrink,
    // 5..10 ct1{b1,b2,be,g,w1,w2}, 11..16 ct2, 17..22 gen{b1,b2,b3,w1,w2,w3},
    // 23..28 gen_c, 29..34 gen_l, 35..40 gen_r, 41 hemiRouting, 42 hemiShrink,
    // 43..48 hs1, 49..54 hs2, 55 localDeepSuper, 56 localRegion, 57 localRouting,
    // 58 localShrink, 59..64 ls1, 65..70 ls2, 71 out, 72 regionDeepSuper,
    // 73 regionHemi, 74 regionRouting, 75 regionShrink, 76..81 rs1, 82..87 rs2

    char* wsb = (char*)d_ws;
    size_t off = 0;
    auto allocB = [&](size_t bytes) -> void* {
        void* p = (void*)(wsb + off);
        off += (bytes + 255) & ~(size_t)255;
        return p;
    };
    auto allocF = [&](size_t n) { return (float*)allocB(n * 4); };

    float* stat = allocF(256);
    float* scl  = allocF(256);
    float* u_ch = allocF((size_t)B * 512);
    float* u_lo = allocF((size_t)B * 256);
    float* u_re = allocF((size_t)B * 128);
    float* u_he = allocF((size_t)B * 64);
    float* v_ch = allocF((size_t)B * 512);
    float* dc   = allocF((size_t)B * 64);
    float* nloc = allocF((size_t)B * 256);
    float* locv = allocF((size_t)B * 256);
    float* dl   = allocF((size_t)B * 128);
    float* lcat = allocF((size_t)B * 512);
    float* lshr = allocF((size_t)B * 512);
    float* nreg = allocF((size_t)B * 256);
    float* regv = allocF((size_t)B * 256);
    float* dr   = allocF((size_t)B * 128);
    float* rcat = allocF((size_t)B * 512);
    float* rshr = allocF((size_t)B * 256);
    float* nhem = allocF((size_t)B * 128);
    float* hemv = allocF((size_t)B * 128);
    float* hcat = allocF((size_t)B * 256);
    float* hshr = allocF((size_t)B * 128);
    float* outv = allocF((size_t)B * 128);
    __bf16* m_o = (__bf16*)allocB((size_t)B * 128 * 2);
    __bf16* m_c = (__bf16*)allocB((size_t)B * 64 * 2);
    __bf16* m_l = (__bf16*)allocB((size_t)B * 128 * 2);
    __bf16* m_r = (__bf16*)allocB((size_t)B * 128 * 2);
    __bf16* h1  = (__bf16*)allocB((size_t)B * 512 * 2);
    __bf16* h2  = (__bf16*)allocB((size_t)B * 1024 * 2);

    // ---- conv branches (2 WMMA passes, BN batch stats in between) ----
    auto run_branch = [&](int base, int G, int KW, int act, float* uout, int coff) {
        const int H = 16 / G, Wout = 192 - KW + 1, C1 = G * 8;
        const int Kreal = H * KW, Kpad = (Kreal + 31) & ~31, ntiles = (Wout + 15) / 16;
        zero_kernel<<<1, 256, 0, stream>>>(stat, 256);
        conv_stats_kernel<<<dim3(B, G), 128, 0, stream>>>(x, F(base + 4), F(base + 0),
                                                          stat, stat + 128,
                                                          G, H, KW, Wout, Kreal, Kpad, ntiles);
        bn_finalize_kernel<<<1, 128, 0, stream>>>(stat, stat + 128, F(base + 3), F(base + 2),
                                                  scl, scl + 128, C1, 1.f / ((float)B * Wout));
        conv2_kernel<<<dim3(B, G), 128, 0, stream>>>(x, F(base + 4), F(base + 0), scl, scl + 128,
                                                     F(base + 5), F(base + 1), uout,
                                                     G, H, KW, Wout, Kreal, Kpad, ntiles,
                                                     act, coff);
    };
    run_branch( 5, 16, 64, 0, u_ch,  0);   // ct1 relu
    run_branch(11, 16, 24, 0, u_ch, 16);   // ct2 relu
    run_branch(59,  8, 36, 0, u_lo,  0);   // ls1 relu
    run_branch(65,  8, 16, 1, u_lo, 16);   // ls2 ELU
    run_branch(76,  4, 24, 0, u_re,  0);   // rs1 relu
    run_branch(82,  4, 32, 0, u_re, 16);   // rs2 relu
    run_branch(43,  2, 30, 0, u_he,  0);   // hs1 relu
    run_branch(49,  2, 60, 0, u_he, 16);   // hs2 relu

    // ---- routing pipeline (templated power-of-2 shapes) ----
    auto concat2 = [&](const float* a, const float* b2_, float* o, int G, int Cc, int E) {
        int total = B * G * 2 * Cc * E;
        concat_caps_kernel<<<(total + 255) / 256, 256, 0, stream>>>(a, b2_, o, B * G, Cc, E);
    };
    routing_kernel< 8, 4,  4,  8><<<B * 16, 32, 0, stream>>>(u_ch, F(3),  v_ch, 16);
    routing_kernel<64, 2,  8, 32><<<B *  1, 32, 0, stream>>>(v_ch, F(2),  dc,    1);
    routing_kernel< 8, 4,  8,  8><<<B *  8, 32, 0, stream>>>(v_ch, F(4),  nloc,  8);
    routing_kernel< 8, 4,  4,  8><<<B *  8, 32, 0, stream>>>(u_lo, F(57), locv,  8);
    routing_kernel<32, 2,  8, 64><<<B *  1, 32, 0, stream>>>(locv, F(55), dl,    1);
    concat2(locv, nloc, lcat, 8, 4, 8);
    routing_kernel< 8, 4,  8, 16><<<B *  8, 32, 0, stream>>>(lcat, F(58), lshr,  8);
    routing_kernel< 8, 4, 16, 16><<<B *  4, 32, 0, stream>>>(lshr, F(56), nreg,  4);
    routing_kernel< 8, 4,  4, 16><<<B *  4, 32, 0, stream>>>(u_re, F(74), regv,  4);
    routing_kernel<16, 2, 16, 64><<<B *  1, 32, 0, stream>>>(regv, F(72), dr,    1);
    concat2(regv, nreg, rcat, 4, 4, 16);
    routing_kernel< 8, 4, 16, 16><<<B *  4, 32, 0, stream>>>(rcat, F(75), rshr,  4);
    routing_kernel< 8, 4, 16, 16><<<B *  2, 32, 0, stream>>>(rshr, F(73), nhem,  2);
    routing_kernel< 8, 4,  4, 16><<<B *  2, 32, 0, stream>>>(u_he, F(41), hemv,  2);
    concat2(hemv, nhem, hcat, 2, 4, 16);
    routing_kernel< 8, 4, 16, 16><<<B *  2, 32, 0, stream>>>(hcat, F(42), hshr,  2);
    routing_kernel< 8, 2, 16, 64><<<B *  1, 32, 0, stream>>>(hshr, F(71), outv,  1);

    // ---- outputs ----
    size_t oPred = 0;
    size_t oRec  = oPred + (size_t)B * 2;
    size_t oLenR = oRec  + (size_t)B * 3072;
    size_t oRecR = oLenR + (size_t)B * 2;
    size_t oLenL = oRecR + (size_t)B * 3072;
    size_t oRecL = oLenL + (size_t)B * 2;
    size_t oLenC = oRecL + (size_t)B * 3072;
    size_t oRecC = oLenC + (size_t)B * 2;

    int g2 = (B * 2 + 255) / 256;
    capslen_kernel<<<g2, 256, 0, stream>>>(outv, dout + oPred, B, 2, 64);
    capslen_kernel<<<g2, 256, 0, stream>>>(dr,   dout + oLenR, B, 2, 64);
    capslen_kernel<<<g2, 256, 0, stream>>>(dl,   dout + oLenL, B, 2, 64);
    capslen_kernel<<<g2, 256, 0, stream>>>(dc,   dout + oLenC, B, 2, 32);

    mask_bf16_kernel<<<(B * 128 + 255) / 256, 256, 0, stream>>>(outv, y, m_o, B, 64);
    mask_bf16_kernel<<<(B * 64  + 255) / 256, 256, 0, stream>>>(dc,   y, m_c, B, 32);
    mask_bf16_kernel<<<(B * 128 + 255) / 256, 256, 0, stream>>>(dl,   y, m_l, B, 64);
    mask_bf16_kernel<<<(B * 128 + 255) / 256, 256, 0, stream>>>(dr,   y, m_r, B, 64);

    // ---- reconstruction MLPs (bf16 WMMA GEMM chain) ----
    auto cvt = [&](const float* s, __bf16* d, size_t n) {
        f32_to_bf16_kernel<<<(int)((n + 1023) / 1024), 256, 0, stream>>>(s, d, (int)n);
    };
    auto run_gen = [&](const __bf16* m, int Kin, int gbase, float* op) {
        __bf16* wa = (__bf16*)allocB((size_t)Kin * 512 * 2);
        __bf16* wb = (__bf16*)allocB((size_t)512 * 1024 * 2);
        __bf16* wc = (__bf16*)allocB((size_t)1024 * 3072 * 2);
        cvt(F(gbase + 3), wa, (size_t)Kin * 512);
        cvt(F(gbase + 4), wb, (size_t)512 * 1024);
        cvt(F(gbase + 5), wc, (size_t)1024 * 3072);
        dim3 blk(256);
        gemm_bf16_kernel<<<dim3((B + BM - 1) / BM, 512 / BN),  blk, 0, stream>>>(
            m,  wa, F(gbase + 0), h1, B, 512,  Kin,  1, 1);
        gemm_bf16_kernel<<<dim3((B + BM - 1) / BM, 1024 / BN), blk, 0, stream>>>(
            h1, wb, F(gbase + 1), h2, B, 1024, 512,  1, 1);
        gemm_bf16_kernel<<<dim3((B + BM - 1) / BM, 3072 / BN), blk, 0, stream>>>(
            h2, wc, F(gbase + 2), op, B, 3072, 1024, 2, 0);
    };
    run_gen(m_o, 128, 17, dout + oRec);    // gen
    run_gen(m_r, 128, 35, dout + oRecR);   // gen_r
    run_gen(m_l, 128, 29, dout + oRecL);   // gen_l
    run_gen(m_c, 64,  23, dout + oRecC);   // gen_c

    (void)n_in; (void)out_size; (void)ws_size;
}
